// GraphAttentionNetwork_75943611728637
// MI455X (gfx1250) — compile-verified, matrix operand order checked by reading, not run on hardware
//
#include <hip/hip_runtime.h>
#include <hip/hip_bf16.h>

#define NNODE 20000
#define NEDGE 160000
#define ETOT  (NEDGE + NNODE)   // self loops appended
#define HEADS 4
#define C1    256
#define HID   1024
#define COUT  512
#define DIN   128

typedef __bf16 bf16_t;
typedef __attribute__((ext_vector_type(16))) __bf16 v16bf;
typedef __attribute__((ext_vector_type(8)))  __bf16 v8bf;
typedef __attribute__((ext_vector_type(8)))  float  v8f;

// ---------------------------------------------------------------- helpers
__device__ __forceinline__ void atomicMaxF(float* addr, float v) {
    if (v >= 0.0f) atomicMax((int*)addr, __float_as_int(v));
    else           atomicMin((unsigned int*)addr, __float_as_uint(v));
}

// ---------------------------------------------------------------- conversions
__global__ __launch_bounds__(256)
void f32_to_bf16_kernel(const float* __restrict__ in, bf16_t* __restrict__ out, size_t n) {
    size_t i = (size_t)blockIdx.x * blockDim.x + threadIdx.x;
    if (i < n) out[i] = (bf16_t)in[i];
}

__global__ __launch_bounds__(256)
void zero_f32_kernel(float* __restrict__ p, size_t n) {
    size_t i = (size_t)blockIdx.x * blockDim.x + threadIdx.x;
    if (i < n) p[i] = 0.0f;
}

// Repack W [K, Ncols] f32 row-major into WMMA-B-fragment-native bf16 layout:
// Bp[(kb*Ncols + n)*32 + pos], K-order within a 32-block: [0..7,16..23,8..15,24..31]
__global__ __launch_bounds__(256)
void repack_B_kernel(const float* __restrict__ W, bf16_t* __restrict__ Bp,
                     int K, int Ncols) {
    size_t idx = (size_t)blockIdx.x * blockDim.x + threadIdx.x;
    size_t total = (size_t)K * Ncols;
    if (idx >= total) return;
    int pos = (int)(idx & 31);
    size_t t = idx >> 5;
    int n  = (int)(t % Ncols);
    int kb = (int)(t / Ncols);
    int koff = (pos < 8) ? pos : (pos < 16) ? pos + 8 : (pos < 24) ? pos - 8 : pos;
    int k = kb * 32 + koff;
    Bp[idx] = (bf16_t)W[(size_t)k * Ncols + n];
}

// ---------------------------------------------------------------- WMMA GEMM
// C[M,Ncols] = A[M,K](bf16, row-major) x Bp(packed bf16). One wave -> 16x64 tile.
__global__ __launch_bounds__(128)
void gemm_bf16_wmma_kernel(const bf16_t* __restrict__ A, const bf16_t* __restrict__ Bp,
                           float* __restrict__ C, int M, int K, int Ncols) {
    const int lane = threadIdx.x & 31;
    const int wave = threadIdx.x >> 5;
    const int m0 = blockIdx.x * 16;
    const int n0 = blockIdx.y * 256 + wave * 64;
    if (m0 >= M || n0 >= Ncols) return;        // wave-uniform

    const int hsel = lane >> 4;                // 0: K{0..7,16..23}  1: K{8..15,24..31}
    const int l16  = lane & 15;

    v8f acc[4];
    for (int q = 0; q < 4; ++q)
        for (int i = 0; i < 8; ++i) acc[q][i] = 0.0f;

    const bf16_t* Arow = A + (size_t)(m0 + l16) * K + hsel * 8;

    for (int k0 = 0; k0 < K; k0 += 32) {
        v16bf afrag;
        {
            v8bf lo = *(const v8bf*)(Arow + k0);
            v8bf hi = *(const v8bf*)(Arow + k0 + 16);
            for (int i = 0; i < 8; ++i) { afrag[i] = lo[i]; afrag[i + 8] = hi[i]; }
        }
        const int kb = k0 >> 5;
#pragma unroll
        for (int q = 0; q < 4; ++q) {
            const int n = n0 + q * 16 + l16;
            v16bf bfrag = *(const v16bf*)(Bp + ((size_t)kb * Ncols + n) * 32 + hsel * 16);
            acc[q] = __builtin_amdgcn_wmma_f32_16x16x32_bf16(
                false, afrag, false, bfrag, (short)0, acc[q], false, false);
        }
    }
    // C/D layout: lanes 0-15 rows M=0..7 in vgpr r, lanes 16-31 rows M=8..15
#pragma unroll
    for (int q = 0; q < 4; ++q) {
        const int n = n0 + q * 16 + l16;
        float* cp = C + (size_t)(m0 + hsel * 8) * Ncols + n;
#pragma unroll
        for (int r = 0; r < 8; ++r) cp[(size_t)r * Ncols] = acc[q][r];
    }
}

// ---------------------------------------------------------------- attention dots
// es[n,h] = sum_c xw[n,h,c]*a_src[h,c];  ed likewise. One wave per (n,h).
__global__ __launch_bounds__(256)
void attn_dots_kernel(const float* __restrict__ xw, const float* __restrict__ a_src,
                      const float* __restrict__ a_dst, float* __restrict__ es,
                      float* __restrict__ ed, int Nn, int Hh, int Cc) {
    int wid  = (int)(((size_t)blockIdx.x * blockDim.x + threadIdx.x) >> 5);
    int lane = threadIdx.x & 31;
    if (wid >= Nn * Hh) return;
    int n = wid / Hh, h = wid % Hh;
    const float* v  = xw + ((size_t)n * Hh + h) * Cc;
    const float* as = a_src + (size_t)h * Cc;
    const float* ad = a_dst + (size_t)h * Cc;
    float s = 0.0f, d = 0.0f;
    for (int c = lane; c < Cc; c += 32) { float xv = v[c]; s += xv * as[c]; d += xv * ad[c]; }
    for (int off = 16; off; off >>= 1) { s += __shfl_xor(s, off, 32); d += __shfl_xor(d, off, 32); }
    if (lane == 0) { es[wid] = s; ed[wid] = d; }
}

__global__ __launch_bounds__(256)
void init_softmax_kernel(float* __restrict__ m, float* __restrict__ denom, int n) {
    int i = blockIdx.x * blockDim.x + threadIdx.x;
    if (i < n) { m[i] = -__builtin_huge_valf(); denom[i] = 0.0f; }
}

// pass 1: e = leaky_relu(es[src]+ed[dst]); store e; segment max into m[dst]
__global__ __launch_bounds__(256)
void edge_max_kernel(const int* __restrict__ ei, const float* __restrict__ es,
                     const float* __restrict__ ed, float* __restrict__ m,
                     float* __restrict__ ebuf, int Hh) {
    int idx = blockIdx.x * blockDim.x + threadIdx.x;
    if (idx >= ETOT * Hh) return;
    int e = idx / Hh, h = idx % Hh;
    int s, dn;
    if (e < NEDGE) { s = ei[e]; dn = ei[NEDGE + e]; } else { s = dn = e - NEDGE; }
    float v = es[s * Hh + h] + ed[dn * Hh + h];
    v = (v > 0.0f) ? v : 0.2f * v;
    ebuf[idx] = v;
    atomicMaxF(&m[dn * Hh + h], v);
}

__global__ __launch_bounds__(256)
void finalize_max_kernel(float* __restrict__ m, int n) {
    int i = blockIdx.x * blockDim.x + threadIdx.x;
    if (i < n) { float v = m[i]; if (!isfinite(v)) m[i] = 0.0f; }
}

// pass 2: p = exp(e - m[dst]); store p; denom[dst] += p
__global__ __launch_bounds__(256)
void edge_exp_kernel(const int* __restrict__ ei, float* __restrict__ pbuf,
                     const float* __restrict__ m, float* __restrict__ denom, int Hh) {
    int idx = blockIdx.x * blockDim.x + threadIdx.x;
    if (idx >= ETOT * Hh) return;
    int e = idx / Hh, h = idx % Hh;
    int dn = (e < NEDGE) ? ei[NEDGE + e] : (e - NEDGE);
    float p = __expf(pbuf[idx] - m[dn * Hh + h]);
    pbuf[idx] = p;
    atomicAdd(&denom[dn * Hh + h], p);
}

// pass 3: agg[dst] += alpha * xw[src]. One wave per edge.
__global__ __launch_bounds__(256)
void edge_aggregate_kernel(const int* __restrict__ ei, const float* __restrict__ pbuf,
                           const float* __restrict__ denom, const float* __restrict__ xw,
                           float* __restrict__ agg, int Hh, int Cc) {
    int wid  = (int)(((size_t)blockIdx.x * blockDim.x + threadIdx.x) >> 5);
    int lane = threadIdx.x & 31;
    if (wid >= ETOT) return;
    int s, dn;
    if (wid < NEDGE) { s = ei[wid]; dn = ei[NEDGE + wid]; } else { s = dn = wid - NEDGE; }
    int HC = Hh * Cc;
    const float* xs = xw + (size_t)s * HC;
    float* od = agg + (size_t)dn * HC;
    for (int h = 0; h < Hh; ++h) {
        float alpha = pbuf[(size_t)wid * Hh + h] / (denom[(size_t)dn * Hh + h] + 1e-16f);
        for (int c = lane; c < Cc; c += 32)
            atomicAdd(&od[h * Cc + c], alpha * xs[h * Cc + c]);
    }
}

// ---------------------------------------------------------------- epilogues
// layers 1/2: y = ELU(LN(agg + bias)*g + b) -> bf16 into next-layer A. D == 1024.
__global__ __launch_bounds__(256)
void bias_ln_elu_kernel(const float* __restrict__ agg, const float* __restrict__ bias,
                        const float* __restrict__ g, const float* __restrict__ bln,
                        bf16_t* __restrict__ outA) {
    const int D = HID;
    __shared__ float red1[8];
    __shared__ float red2[8];
    int n = blockIdx.x;
    const float* row = agg + (size_t)n * D;
    float v[4]; float s = 0.0f, sq = 0.0f;
#pragma unroll
    for (int i = 0; i < 4; ++i) {
        int c = threadIdx.x + i * 256;
        float t = row[c] + bias[c];
        v[i] = t; s += t; sq += t * t;
    }
    for (int off = 16; off; off >>= 1) { s += __shfl_xor(s, off, 32); sq += __shfl_xor(sq, off, 32); }
    int w = threadIdx.x >> 5;
    if ((threadIdx.x & 31) == 0) { red1[w] = s; red2[w] = sq; }
    __syncthreads();
    float ts = 0.0f, tq = 0.0f;
    for (int i = 0; i < 8; ++i) { ts += red1[i]; tq += red2[i]; }
    float mu  = ts / (float)D;
    float var = tq / (float)D - mu * mu;
    float inv = rsqrtf(var + 1e-5f);
#pragma unroll
    for (int i = 0; i < 4; ++i) {
        int c = threadIdx.x + i * 256;
        float t = (v[i] - mu) * inv * g[c] + bln[c];
        t = (t > 0.0f) ? t : (__expf(t) - 1.0f);
        outA[(size_t)n * D + c] = (bf16_t)t;
    }
}

// layer 3: out += b3 (in place on d_out)
__global__ __launch_bounds__(256)
void bias_add_kernel(float* __restrict__ out, const float* __restrict__ bias, int HC, size_t n) {
    size_t i = (size_t)blockIdx.x * blockDim.x + threadIdx.x;
    if (i < n) out[i] += bias[i % HC];
}

// ---------------------------------------------------------------- launch
extern "C" void kernel_launch(void* const* d_in, const int* in_sizes, int n_in,
                              void* d_out, int out_size, void* d_ws, size_t ws_size,
                              hipStream_t stream) {
    const float* x      = (const float*)d_in[0];
    const int*   ei     = (const int*)d_in[1];
    const float* W1     = (const float*)d_in[2];
    const float* a_src1 = (const float*)d_in[3];
    const float* a_dst1 = (const float*)d_in[4];
    const float* b1     = (const float*)d_in[5];
    const float* ln1_g  = (const float*)d_in[6];
    const float* ln1_b  = (const float*)d_in[7];
    const float* W2     = (const float*)d_in[8];
    const float* a_src2 = (const float*)d_in[9];
    const float* a_dst2 = (const float*)d_in[10];
    const float* b2     = (const float*)d_in[11];
    const float* ln2_g  = (const float*)d_in[12];
    const float* ln2_b  = (const float*)d_in[13];
    const float* W3     = (const float*)d_in[14];
    const float* a_src3 = (const float*)d_in[15];
    const float* a_dst3 = (const float*)d_in[16];
    const float* b3     = (const float*)d_in[17];
    float* out = (float*)d_out;

    // workspace carve (256B aligned)
    char* wp = (char*)d_ws;
    auto carve = [&](size_t bytes) -> char* {
        char* p = wp; wp += (bytes + 255) & ~(size_t)255; return p;
    };
    bf16_t* Ab    = (bf16_t*)carve((size_t)NNODE * HID * sizeof(bf16_t));
    bf16_t* Bp    = (bf16_t*)carve((size_t)HID * HID * sizeof(bf16_t));
    float*  xw    = (float*)carve((size_t)NNODE * HID * sizeof(float));
    float*  agg   = (float*)carve((size_t)NNODE * HID * sizeof(float));
    float*  es    = (float*)carve((size_t)NNODE * HEADS * sizeof(float));
    float*  ed    = (float*)carve((size_t)NNODE * HEADS * sizeof(float));
    float*  mbuf  = (float*)carve((size_t)NNODE * HEADS * sizeof(float));
    float*  denom = (float*)carve((size_t)NNODE * HEADS * sizeof(float));
    float*  pbuf  = (float*)carve((size_t)ETOT * HEADS * sizeof(float));

    auto cdiv = [](size_t a, size_t b) { return (unsigned)((a + b - 1) / b); };

    auto run_gat = [&](int K, int Hh, int Cc,
                       const float* W, const float* asrc, const float* adst,
                       const float* bias, const float* lng, const float* lnb,
                       float* aggdst, bool last) {
        int HC = Hh * Cc;
        size_t wtot = (size_t)K * HC;
        repack_B_kernel<<<cdiv(wtot, 256), 256, 0, stream>>>(W, Bp, K, HC);
        gemm_bf16_wmma_kernel<<<dim3(NNODE / 16, HC / 256), 128, 0, stream>>>(
            Ab, Bp, xw, NNODE, K, HC);
        int pairs = NNODE * Hh;
        attn_dots_kernel<<<cdiv((size_t)pairs * 32, 256), 256, 0, stream>>>(
            xw, asrc, adst, es, ed, NNODE, Hh, Cc);
        init_softmax_kernel<<<cdiv(pairs, 256), 256, 0, stream>>>(mbuf, denom, pairs);
        edge_max_kernel<<<cdiv((size_t)ETOT * Hh, 256), 256, 0, stream>>>(
            ei, es, ed, mbuf, pbuf, Hh);
        finalize_max_kernel<<<cdiv(pairs, 256), 256, 0, stream>>>(mbuf, pairs);
        edge_exp_kernel<<<cdiv((size_t)ETOT * Hh, 256), 256, 0, stream>>>(
            ei, pbuf, mbuf, denom, Hh);
        size_t aggN = (size_t)NNODE * HC;
        zero_f32_kernel<<<cdiv(aggN, 256), 256, 0, stream>>>(aggdst, aggN);
        edge_aggregate_kernel<<<cdiv((size_t)ETOT * 32, 256), 256, 0, stream>>>(
            ei, pbuf, denom, xw, aggdst, Hh, Cc);
        if (!last)
            bias_ln_elu_kernel<<<NNODE, 256, 0, stream>>>(aggdst, bias, lng, lnb, Ab);
        else
            bias_add_kernel<<<cdiv(aggN, 256), 256, 0, stream>>>(aggdst, bias, HC, aggN);
    };

    // x -> bf16 A for layer 1
    f32_to_bf16_kernel<<<cdiv((size_t)NNODE * DIN, 256), 256, 0, stream>>>(
        x, Ab, (size_t)NNODE * DIN);

    run_gat(DIN, HEADS, C1, W1, a_src1, a_dst1, b1, ln1_g, ln1_b, agg, false);
    run_gat(HID, HEADS, C1, W2, a_src2, a_dst2, b2, ln2_g, ln2_b, agg, false);
    run_gat(HID, 1, COUT, W3, a_src3, a_dst3, b3, nullptr, nullptr, out, true);
}
